// HybridLHM_28089086116624
// MI455X (gfx1250) — matrix operand order repaired and checked by reading, not compiled, not run on hardware
//
#include <hip/hip_runtime.h>
#include <hip/hip_bf16.h>
#include <math.h>

// ---- model dims ----
#define DD   128
#define LL   2048
#define BB   4
#define DI   256
#define DS   16
#define KK   4
#define HH   4
#define NM   6
#define NA   2
#define NTOK (BB*LL)          // 8192 tokens

typedef __attribute__((ext_vector_type(16))) _Float16 v16h;
typedef __attribute__((ext_vector_type(8)))  float    v8f;
typedef __attribute__((ext_vector_type(4)))  _Float16 h4;

// ---- activations ----
#define ACT_NONE     0
#define ACT_SOFTPLUS 1
#define ACT_GELU     2

__device__ __forceinline__ float apply_act(float v, int act) {
  if (act == ACT_SOFTPLUS) {                       // log1p(exp(x)), overflow-safe
    return (v > 20.f) ? v : log1pf(expf(v));
  } else if (act == ACT_GELU) {                    // exact GELU
    return 0.5f * v * (1.f + erff(v * 0.70710678118654752f));
  }
  return v;
}

// =====================================================================
// Tiled WMMA GEMM:  C[M,Nout] = act(A[M,Kd] @ W[Nout,Kd]^T + bias) (+Res)
// BM=BN=64, BK=32, 256 threads (8 waves); each wave: two 16x16 C tiles.
// Double-buffered LDS: tile i+1 prefetched into registers while WMMAs
// consume tile i; single s_barrier per K-step. GUARD template removes
// all bounds checks when Nout % 64 == 0 (every GEMM except the DS=16
// B/C projections).
// =====================================================================
#define BM 64
#define BN 64
#define BK 32
#define LDP (BK + 8)          // LDS row stride in halfs: 40 halfs = 80B (8B aligned)

template <bool GUARD>
__global__ void __launch_bounds__(256)
gemm_wmma_kernel(const float* __restrict__ A, const float* __restrict__ W,
                 const float* __restrict__ bias, const float* __restrict__ Res,
                 float* __restrict__ C, int M, int Nout, int Kd, int act) {
  __shared__ _Float16 As[2][BM][LDP];
  __shared__ _Float16 Ws[2][BN][LDP];

  const int tid  = threadIdx.x;
  const int wave = tid >> 5;
  const int lane = tid & 31;
  const int wr   = wave >> 2;        // 0..1  (M sub-block of 32)
  const int wc   = wave & 3;         // 0..3  (N sub-block of 16)
  const int hi   = lane >> 4;        // lane half
  const int r    = lane & 15;

  const int m0 = blockIdx.x * BM;
  const int n0 = blockIdx.y * BN;

  // staging coordinates: each thread moves 2 float4 of A and 2 of W per step
  const int r0 = tid >> 3;           // 0..31
  const int c4 = (tid & 7) * 4;      // 0,4,...,28
  const int wn0 = n0 + r0;
  const int wn1 = n0 + r0 + 32;

  auto load_tile = [&](int k0, float4& a0v, float4& a1v, float4& w0v, float4& w1v) {
    const float* Ap = A + (size_t)(m0 + r0) * Kd + k0 + c4;
    a0v = *(const float4*)Ap;
    a1v = *(const float4*)(Ap + (size_t)32 * Kd);
    if (GUARD) {
      w0v = make_float4(0.f, 0.f, 0.f, 0.f);
      w1v = make_float4(0.f, 0.f, 0.f, 0.f);
      if (wn0 < Nout) w0v = *(const float4*)(W + (size_t)wn0 * Kd + k0 + c4);
      if (wn1 < Nout) w1v = *(const float4*)(W + (size_t)wn1 * Kd + k0 + c4);
    } else {
      w0v = *(const float4*)(W + (size_t)wn0 * Kd + k0 + c4);
      w1v = *(const float4*)(W + (size_t)wn1 * Kd + k0 + c4);
    }
  };
  auto store_tile = [&](int buf, float4 a0v, float4 a1v, float4 w0v, float4 w1v) {
    h4 ah0 = { (_Float16)a0v.x, (_Float16)a0v.y, (_Float16)a0v.z, (_Float16)a0v.w };
    h4 ah1 = { (_Float16)a1v.x, (_Float16)a1v.y, (_Float16)a1v.z, (_Float16)a1v.w };
    h4 wh0 = { (_Float16)w0v.x, (_Float16)w0v.y, (_Float16)w0v.z, (_Float16)w0v.w };
    h4 wh1 = { (_Float16)w1v.x, (_Float16)w1v.y, (_Float16)w1v.z, (_Float16)w1v.w };
    *(h4*)&As[buf][r0][c4]      = ah0;
    *(h4*)&As[buf][r0 + 32][c4] = ah1;
    *(h4*)&Ws[buf][r0][c4]      = wh0;
    *(h4*)&Ws[buf][r0 + 32][c4] = wh1;
  };

  v8f c0 = {}; v8f c1 = {};

  float4 a0v, a1v, w0v, w1v;
  load_tile(0, a0v, a1v, w0v, w1v);
  store_tile(0, a0v, a1v, w0v, w1v);
  __syncthreads();

  const int NIT = Kd / BK;
  for (int it = 0; it < NIT; ++it) {
    const int buf  = it & 1;
    const bool more = (it + 1 < NIT);
    // prefetch next tile into registers (overlaps with WMMA below)
    if (more) load_tile((it + 1) * BK, a0v, a1v, w0v, w1v);

    // fragments, contiguous k = hi*16 + e in both A and B (16-bit A 16x32, B 32x16)
    v16h a0 = {}, a1 = {}, bf = {};
#pragma unroll
    for (int e = 0; e < 16; ++e) {
      int kk = hi * 16 + e;
      a0[e] = As[buf][wr * 32 + r][kk];
      a1[e] = As[buf][wr * 32 + 16 + r][kk];
      bf[e] = Ws[buf][wc * 16 + r][hi * 16 + e];
    }
    c0 = __builtin_amdgcn_wmma_f32_16x16x32_f16(false, a0, false, bf, (short)0, c0, false, false);
    c1 = __builtin_amdgcn_wmma_f32_16x16x32_f16(false, a1, false, bf, (short)0, c1, false, false);

    if (more) {
      store_tile(buf ^ 1, a0v, a1v, w0v, w1v);
      __syncthreads();
    }
  }

  // epilogue: bias + activation + optional residual
  const int col = n0 + wc * 16 + r;
  if (!GUARD || col < Nout) {
    float bv = bias ? bias[col] : 0.f;
#pragma unroll
    for (int r8 = 0; r8 < 8; ++r8) {
      int row0 = m0 + wr * 32 + r8 + hi * 8;      // c0 tile rows
      int row1 = row0 + 16;                        // c1 tile rows
      float v0 = apply_act(c0[r8] + bv, act);
      float v1 = apply_act(c1[r8] + bv, act);
      if (Res) {
        v0 += Res[(size_t)row0 * Nout + col];
        v1 += Res[(size_t)row1 * Nout + col];
      }
      C[(size_t)row0 * Nout + col] = v0;
      C[(size_t)row1 * Nout + col] = v1;
    }
  }
}

// =====================================================================
// Embedding: token emb + type emb + time encoding @ te_w^T + te_b
// =====================================================================
__global__ void __launch_bounds__(128)
embed_kernel(const int* __restrict__ ids, const int* __restrict__ tts,
             const float* __restrict__ ts, const float* __restrict__ emb,
             const float* __restrict__ temb, const float* __restrict__ freqs,
             const float* __restrict__ phases, const float* __restrict__ te_w,
             const float* __restrict__ te_b, float* __restrict__ x) {
  __shared__ float enc[DD];
  int t = blockIdx.x;
  int d = threadIdx.x;
  float tv = ts[t];
  if (d < DD / 2) enc[d] = sinf(tv * freqs[d] + phases[d]);
  else            enc[d] = cosf(tv * freqs[d - DD / 2] + phases[d - DD / 2]);
  __syncthreads();
  int id = ids[t], tt = tts[t];
  float acc = te_b[d] + emb[(size_t)id * DD + d] + temb[(size_t)tt * DD + d];
  const float* wr = te_w + (size_t)d * DD;
#pragma unroll 8
  for (int j = 0; j < DD; ++j) acc += wr[j] * enc[j];
  x[(size_t)t * DD + d] = acc;
}

// ---- LayerNorm over D=128; one block per token ----
__global__ void __launch_bounds__(128)
ln_kernel(const float* __restrict__ x, const float* __restrict__ w,
          const float* __restrict__ b, float* __restrict__ out) {
  __shared__ float red[DD];
  int t = blockIdx.x, d = threadIdx.x;
  float v = x[(size_t)t * DD + d];
  red[d] = v; __syncthreads();
  for (int s = 64; s > 0; s >>= 1) { if (d < s) red[d] += red[d + s]; __syncthreads(); }
  float mu = red[0] * (1.f / DD); __syncthreads();
  float df = v - mu;
  red[d] = df * df; __syncthreads();
  for (int s = 64; s > 0; s >>= 1) { if (d < s) red[d] += red[d + s]; __syncthreads(); }
  float var = red[0] * (1.f / DD);
  out[(size_t)t * DD + d] = df * rsqrtf(var + 1e-5f) * w[d] + b[d];
}

// ---- depthwise causal conv (K=4) + SiLU ; xp = xz[:, :DI] ----
__global__ void __launch_bounds__(256)
conv_silu_kernel(const float* __restrict__ xz, const float* __restrict__ cw,
                 const float* __restrict__ cb, float* __restrict__ xc) {
  int idx = blockIdx.x * blockDim.x + threadIdx.x;   // over NTOK*DI
  int c = idx & (DI - 1);
  int t = idx >> 8;                                  // token index (DI=256)
  int l = t & (LL - 1);                              // position in sequence
  float acc = cb[c];
#pragma unroll
  for (int k = 0; k < KK; ++k) {
    int ls = l - (KK - 1) + k;
    if (ls >= 0) acc += cw[c * KK + k] * xz[(size_t)(t - (KK - 1) + k) * (2 * DI) + c];
  }
  xc[(size_t)t * DI + c] = acc / (1.f + expf(-acc)); // silu
}

// ---- selective scan: one lane per (batch, channel); DS=16 states ----
__global__ void __launch_bounds__(256)
scan_kernel(const float* __restrict__ dt, const float* __restrict__ xc,
            const float* __restrict__ Bm, const float* __restrict__ Cm,
            const float* __restrict__ Alog, float* __restrict__ y) {
  int idx = blockIdx.x * blockDim.x + threadIdx.x;   // b*DI + c
  int c = idx & (DI - 1);
  int b = idx >> 8;
  float A[DS], h[DS];
#pragma unroll
  for (int s = 0; s < DS; ++s) { A[s] = -__expf(Alog[c * DS + s]); h[s] = 0.f; }
  for (int l = 0; l < LL; ++l) {
    size_t t = (size_t)b * LL + l;
    float dtv = dt[t * DI + c];
    float xv  = xc[t * DI + c];
    float xd  = xv * dtv;
    const float* bp = Bm + t * DS;
    const float* cp = Cm + t * DS;
    float acc = 0.f;
#pragma unroll
    for (int s = 0; s < DS; ++s) {
      h[s] = h[s] * __expf(A[s] * dtv) + xd * bp[s];
      acc += h[s] * cp[s];
    }
    y[t * DI + c] = acc;
  }
}

// ---- gating: y = (y + xc*Dp) * silu(z), z = xz[:, DI + c] ----
__global__ void __launch_bounds__(256)
gate_kernel(float* __restrict__ y, const float* __restrict__ xc,
            const float* __restrict__ xz, const float* __restrict__ Dp) {
  int idx = blockIdx.x * blockDim.x + threadIdx.x;
  int c = idx & (DI - 1);
  size_t t = (size_t)(idx >> 8);
  float z  = xz[t * (2 * DI) + DI + c];
  float sz = z / (1.f + expf(-z));
  y[idx] = (y[idx] + xc[idx] * Dp[c]) * sz;
}

// =====================================================================
// WMMA flash attention. dh=32 == WMMA K. Block = one (b,h,16-query tile),
// 8 waves; each wave owns 32-key chunks strided by 8. Next chunk's K/V
// rows are prefetched (global_prefetch) while the current chunk's WMMAs
// run. Final flash-combine of 8 wave-partials through LDS.
// =====================================================================
#define ATW 8
#define VLP 40                 // LDS row stride (halfs) for P/V tiles

__global__ void __launch_bounds__(256)
attn_wmma_kernel(const float* __restrict__ qkv, float* __restrict__ attno) {
  __shared__ _Float16 Qs[16][32];
  __shared__ _Float16 Pls[ATW][16][VLP];
  __shared__ _Float16 Vls[ATW][32][VLP];
  __shared__ float Om[ATW][16][32];
  __shared__ float Mrow[ATW][16];
  __shared__ float Lrow[ATW][16];

  const int tid  = threadIdx.x;
  const int wave = tid >> 5;
  const int lane = tid & 31;
  const int hi   = lane >> 4;
  const int r    = lane & 15;

  const int qt = blockIdx.x & (LL / 16 - 1);     // 128 query tiles
  const int bh = blockIdx.x >> 7;
  const int h  = bh & (HH - 1);
  const int b  = bh >> 2;
  const int q0 = qt * 16;
  const int QW = 3 * DD;
  const float scale = 0.17677669529663687f;      // 1/sqrt(32)
  const size_t base = (size_t)b * LL;

  // stage scaled Q (16x32)
  for (int t = tid; t < 16 * 32; t += 256) {
    int row = t >> 5, d = t & 31;
    Qs[row][d] = (_Float16)(qkv[(base + q0 + row) * QW + h * 32 + d] * scale);
  }
  __syncthreads();

  // Q as A-fragment (rows = queries, K = dh)
  v16h qf = {};
#pragma unroll
  for (int e = 0; e < 16; ++e) {
    int kk = hi * 16 + e;
    qf[e] = Qs[r][kk];
  }

  v8f Oa = {}, Ob = {};
  float mrw[8], lrw[8];
#pragma unroll
  for (int p = 0; p < 8; ++p) { mrw[p] = -3.4e38f; lrw[p] = 0.f; }

  const int NCH = LL / 32;                       // 64 key chunks
  for (int ch = wave; ch < NCH; ch += ATW) {
    const int kbase = ch * 32;

    // prefetch next chunk's K/V rows for this wave (global_prefetch path)
    const int chn = ch + ATW;
    if (chn < NCH) {
      const float* nk = qkv + (base + chn * 32 + lane) * QW + DD + h * 32;
      const float* nv = qkv + (base + chn * 32 + lane) * QW + 2 * DD + h * 32;
      __builtin_prefetch(nk, 0, 1);
      __builtin_prefetch(nv, 0, 1);
    }

    // K^T B-fragments: col = key (r), K = dh (hi*16+e); 64B contiguous per lane
    v16h kb0 = {}, kb1 = {};
    {
      const float* kp0 = qkv + (base + kbase + r) * QW + DD + h * 32 + hi * 16;
      const float* kp1 = qkv + (base + kbase + 16 + r) * QW + DD + h * 32 + hi * 16;
#pragma unroll
      for (int e = 0; e < 16; ++e) {
        kb0[e] = (_Float16)kp0[e];
        kb1[e] = (_Float16)kp1[e];
      }
    }
    // stage this wave's V tile (32 keys x 32 dh), contiguous per lane
    {
      const float* vp = qkv + (base + kbase + lane) * QW + 2 * DD + h * 32;
#pragma unroll
      for (int d = 0; d < 32; ++d) Vls[wave][lane][d] = (_Float16)vp[d];
    }

    // scores: 16 queries x 32 keys
    v8f S0 = {}, S1 = {};
    S0 = __builtin_amdgcn_wmma_f32_16x16x32_f16(false, qf, false, kb0, (short)0, S0, false, false);
    S1 = __builtin_amdgcn_wmma_f32_16x16x32_f16(false, qf, false, kb1, (short)0, S1, false, false);

    // online softmax per row (row = hi*8+p; values across 16 lanes of half)
    float mx[8];
#pragma unroll
    for (int p = 0; p < 8; ++p) mx[p] = fmaxf(S0[p], S1[p]);
#pragma unroll
    for (int off = 8; off > 0; off >>= 1) {
#pragma unroll
      for (int p = 0; p < 8; ++p) mx[p] = fmaxf(mx[p], __shfl_xor(mx[p], off, 32));
    }

    float P0[8], P1[8], rs[8];
#pragma unroll
    for (int p = 0; p < 8; ++p) {
      float mn = fmaxf(mrw[p], mx[p]);
      float sc = __expf(mrw[p] - mn);
      mrw[p] = mn;
      P0[p] = __expf(S0[p] - mn);
      P1[p] = __expf(S1[p] - mn);
      rs[p] = P0[p] + P1[p];
      lrw[p] *= sc;
      Oa[p] *= sc;
      Ob[p] *= sc;
    }
#pragma unroll
    for (int off = 8; off > 0; off >>= 1) {
#pragma unroll
      for (int p = 0; p < 8; ++p) rs[p] += __shfl_xor(rs[p], off, 32);
    }
#pragma unroll
    for (int p = 0; p < 8; ++p) lrw[p] += rs[p];

    // C-layout -> A-layout transpose of P through per-wave LDS
#pragma unroll
    for (int p = 0; p < 8; ++p) {
      Pls[wave][hi * 8 + p][r]      = (_Float16)P0[p];
      Pls[wave][hi * 8 + p][16 + r] = (_Float16)P1[p];
    }
    v16h pf = {}, va = {}, vb = {};
#pragma unroll
    for (int e = 0; e < 16; ++e) {
      int kk = hi * 16 + e;
      pf[e] = Pls[wave][r][kk];
      va[e] = Vls[wave][hi * 16 + e][r];        // B-frag: col=dh r, K=key
      vb[e] = Vls[wave][hi * 16 + e][16 + r];
    }
    Oa = __builtin_amdgcn_wmma_f32_16x16x32_f16(false, pf, false, va, (short)0, Oa, false, false);
    Ob = __builtin_amdgcn_wmma_f32_16x16x32_f16(false, pf, false, vb, (short)0, Ob, false, false);
  }

  // publish per-wave partials
#pragma unroll
  for (int p = 0; p < 8; ++p) {
    Om[wave][hi * 8 + p][r]      = Oa[p];
    Om[wave][hi * 8 + p][16 + r] = Ob[p];
  }
  if (r == 0) {
#pragma unroll
    for (int p = 0; p < 8; ++p) {
      Mrow[wave][hi * 8 + p] = mrw[p];
      Lrow[wave][hi * 8 + p] = lrw[p];
    }
  }
  __syncthreads();

  // flash-combine 8 partials; 512 outputs over 256 threads
  for (int t = tid; t < 16 * 32; t += 256) {
    int row = t >> 5, d = t & 31;
    float M = -3.4e38f;
#pragma unroll
    for (int w = 0; w < ATW; ++w) M = fmaxf(M, Mrow[w][row]);
    float Ls = 0.f, Ov = 0.f;
#pragma unroll
    for (int w = 0; w < ATW; ++w) {
      float f = __expf(Mrow[w][row] - M);
      Ls += Lrow[w][row] * f;
      Ov += Om[w][row][d] * f;
    }
    attno[(base + q0 + row) * DD + h * 32 + d] = Ov / Ls;
  }
}

// ---- head: LN(cls) @ h_w^T + h_b ; one block per batch ----
__global__ void __launch_bounds__(128)
head_kernel(const float* __restrict__ x, const float* __restrict__ w,
            const float* __restrict__ b, const float* __restrict__ hw,
            const float* __restrict__ hb, float* __restrict__ out) {
  __shared__ float red[DD];
  int bb = blockIdx.x, d = threadIdx.x;
  float v = x[(size_t)bb * LL * DD + d];
  red[d] = v; __syncthreads();
  for (int s = 64; s > 0; s >>= 1) { if (d < s) red[d] += red[d + s]; __syncthreads(); }
  float mu = red[0] * (1.f / DD); __syncthreads();
  float df = v - mu;
  red[d] = df * df; __syncthreads();
  for (int s = 64; s > 0; s >>= 1) { if (d < s) red[d] += red[d + s]; __syncthreads(); }
  float var = red[0] * (1.f / DD); __syncthreads();
  float ln = df * rsqrtf(var + 1e-5f) * w[d] + b[d];
  red[d] = ln * hw[d]; __syncthreads();
  for (int s = 64; s > 0; s >>= 1) { if (d < s) red[d] += red[d + s]; __syncthreads(); }
  if (d == 0) out[bb] = red[0] + hb[0];
}

// =====================================================================
// host-side orchestration
// =====================================================================
static inline void launch_gemm(const float* A, const float* W, const float* bias,
                               const float* Res, float* C, int M, int Nout, int Kd,
                               int act, hipStream_t s) {
  dim3 grid(M / BM, (Nout + BN - 1) / BN);
  if ((Nout % BN) == 0)
    gemm_wmma_kernel<false><<<grid, 256, 0, s>>>(A, W, bias, Res, C, M, Nout, Kd, act);
  else
    gemm_wmma_kernel<true><<<grid, 256, 0, s>>>(A, W, bias, Res, C, M, Nout, Kd, act);
}

extern "C" void kernel_launch(void* const* d_in, const int* in_sizes, int n_in,
                              void* d_out, int out_size, void* d_ws, size_t ws_size,
                              hipStream_t stream) {
  (void)in_sizes; (void)n_in; (void)out_size; (void)ws_size;
  const int*   ids      = (const int*)  d_in[0];
  const int*   tts      = (const int*)  d_in[1];
  const float* ts       = (const float*)d_in[2];
  const float* emb      = (const float*)d_in[3];
  const float* temb     = (const float*)d_in[4];
  const float* freqs    = (const float*)d_in[5];
  const float* phases   = (const float*)d_in[6];
  const float* te_w     = (const float*)d_in[7];
  const float* te_b     = (const float*)d_in[8];
  const float* m_norm_w = (const float*)d_in[9];
  const float* m_norm_b = (const float*)d_in[10];
  const float* m_in_w   = (const float*)d_in[11];
  const float* m_conv_w = (const float*)d_in[12];
  const float* m_conv_b = (const float*)d_in[13];
  const float* m_dt_w   = (const float*)d_in[14];
  const float* m_dt_b   = (const float*)d_in[15];
  const float* m_B_w    = (const float*)d_in[16];
  const float* m_C_w    = (const float*)d_in[17];
  const float* m_D      = (const float*)d_in[18];
  const float* m_A_log  = (const float*)d_in[19];
  const float* m_out_w  = (const float*)d_in[20];
  const float* a_ln1_w  = (const float*)d_in[21];
  const float* a_ln1_b  = (const float*)d_in[22];
  const float* a_qkv_w  = (const float*)d_in[23];
  const float* a_qkv_b  = (const float*)d_in[24];
  const float* a_out_w  = (const float*)d_in[25];
  const float* a_out_b  = (const float*)d_in[26];
  const float* a_ln2_w  = (const float*)d_in[27];
  const float* a_ln2_b  = (const float*)d_in[28];
  const float* a_f1_w   = (const float*)d_in[29];
  const float* a_f1_b   = (const float*)d_in[30];
  const float* a_f2_w   = (const float*)d_in[31];
  const float* a_f2_b   = (const float*)d_in[32];
  const float* h_ln_w   = (const float*)d_in[33];
  const float* h_ln_b   = (const float*)d_in[34];
  const float* h_w      = (const float*)d_in[35];
  const float* h_b      = (const float*)d_in[36];
  float* out = (float*)d_out;

  // workspace layout (floats), ~68MB total
  float* ws  = (float*)d_ws;
  const size_t N = NTOK;
  float* x   = ws;                         // N*D
  float* xn  = x   + N * DD;               // N*D
  float* xz  = xn  + N * DD;               // N*2DI (also reused as FFN hidden)
  float* xc  = xz  + N * 2 * DI;           // N*DI
  float* dtb = xc  + N * DI;               // N*DI
  float* Bm  = dtb + N * DI;               // N*DS
  float* Cm  = Bm  + N * DS;               // N*DS
  float* yb  = Cm  + N * DS;               // N*DI
  float* qkv = yb  + N * DI;               // N*3D
  float* ao  = qkv + N * 3 * DD;           // N*D

  embed_kernel<<<N, 128, 0, stream>>>(ids, tts, ts, emb, temb, freqs, phases,
                                      te_w, te_b, x);

  const int attn_every = (NM + NA) / NA;   // 4
  int mi = 0, ai = 0;
  for (int i = 0; i < NM + NA; ++i) {
    if ((i + 1) % attn_every == 0) {
      // ---- attention block ai ----
      ln_kernel<<<N, 128, 0, stream>>>(x, a_ln1_w + ai * DD, a_ln1_b + ai * DD, xn);
      launch_gemm(xn, a_qkv_w + (size_t)ai * 3 * DD * DD, a_qkv_b + ai * 3 * DD,
                  nullptr, qkv, N, 3 * DD, DD, ACT_NONE, stream);
      attn_wmma_kernel<<<BB * HH * (LL / 16), 256, 0, stream>>>(qkv, ao);
      launch_gemm(ao, a_out_w + (size_t)ai * DD * DD, a_out_b + ai * DD,
                  x, x, N, DD, DD, ACT_NONE, stream);
      ln_kernel<<<N, 128, 0, stream>>>(x, a_ln2_w + ai * DD, a_ln2_b + ai * DD, xn);
      launch_gemm(xn, a_f1_w + (size_t)ai * 4 * DD * DD, a_f1_b + ai * 4 * DD,
                  nullptr, xz, N, 4 * DD, DD, ACT_GELU, stream);
      launch_gemm(xz, a_f2_w + (size_t)ai * DD * 4 * DD, a_f2_b + ai * DD,
                  x, x, N, DD, 4 * DD, ACT_NONE, stream);
      ++ai;
    } else {
      // ---- mamba block mi ----
      ln_kernel<<<N, 128, 0, stream>>>(x, m_norm_w + mi * DD, m_norm_b + mi * DD, xn);
      launch_gemm(xn, m_in_w + (size_t)mi * 2 * DI * DD, nullptr,
                  nullptr, xz, N, 2 * DI, DD, ACT_NONE, stream);
      conv_silu_kernel<<<(N * DI) / 256, 256, 0, stream>>>(
          xz, m_conv_w + (size_t)mi * DI * KK, m_conv_b + mi * DI, xc);
      launch_gemm(xc, m_dt_w + (size_t)mi * DI * DI, m_dt_b + mi * DI,
                  nullptr, dtb, N, DI, DI, ACT_SOFTPLUS, stream);
      launch_gemm(xc, m_B_w + (size_t)mi * DS * DI, nullptr,
                  nullptr, Bm, N, DS, DI, ACT_NONE, stream);
      launch_gemm(xc, m_C_w + (size_t)mi * DS * DI, nullptr,
                  nullptr, Cm, N, DS, DI, ACT_NONE, stream);
      scan_kernel<<<(BB * DI) / 256, 256, 0, stream>>>(
          dtb, xc, Bm, Cm, m_A_log + (size_t)mi * DI * DS, yb);
      gate_kernel<<<(N * DI) / 256, 256, 0, stream>>>(yb, xc, xz, m_D + mi * DI);
      launch_gemm(yb, m_out_w + (size_t)mi * DD * DI, nullptr,
                  x, x, N, DD, DI, ACT_NONE, stream);
      ++mi;
    }
  }

  head_kernel<<<BB, 128, 0, stream>>>(x, h_ln_w, h_ln_b, h_w, h_b, out);
}
